// VisionMamba_83760452206811
// MI455X (gfx1250) — matrix operand order repaired and checked by reading, not compiled, not run on hardware
//
#include <hip/hip_runtime.h>
#include <hip/hip_bf16.h>

#define DEV __device__ __forceinline__

typedef __attribute__((ext_vector_type(16))) _Float16 v16h;
typedef __attribute__((ext_vector_type(8)))  _Float16 v8h;
typedef __attribute__((ext_vector_type(8)))  float    v8f;
typedef unsigned v4u __attribute__((ext_vector_type(4)));
typedef int      v8i __attribute__((ext_vector_type(8)));
typedef int      v4i __attribute__((ext_vector_type(4)));

#if __has_builtin(__builtin_amdgcn_tensor_load_to_lds)
#define HAS_TDM 1
#else
#define HAS_TDM 0
#endif

// ---------------- model constants ----------------
constexpr int BATCH = 16;
constexpr int L     = 1024;
constexpr int M     = BATCH * L;   // 16384 GEMM rows
constexpr int E     = 192;
constexpr int DM    = 196;         // D_MODEL
constexpr int DI    = 392;         // D_INNER
constexpr int DTR   = 13;          // DT_RANK
constexpr int DS    = 16;          // D_STATE
// padded dims (K padded to mult of 32, N to mult of 64)
constexpr int KP_H    = 224;  // 196 -> 224
constexpr int NP_XZ   = 832;  // 784 -> 832
constexpr int KP_XC   = 416;  // 392 -> 416
constexpr int NP_XDBL = 64;   // 45  -> 64
constexpr int KP_DT   = 32;   // 13  -> 32
constexpr int NP_DLT  = 448;  // 392 -> 448
constexpr int NP_OUT  = 256;  // 196 -> 256

DEV float silu_f(float x)     { return x / (1.f + __expf(-x)); }
DEV float softplus_f(float x) { return x > 20.f ? x : log1pf(__expf(x)); }

DEV v16h cat8(v8h lo, v8h hi) {
  v16h r;
#pragma unroll
  for (int i = 0; i < 8; ++i) { r[i] = lo[i]; r[8 + i] = hi[i]; }
  return r;
}

// ---------------- generic WMMA f16 GEMM ----------------
// C[M][Np] = act(A[M][Kp] * B[Np][Kp]^T + bias) (+ beta*C)
// A tiles staged with GLOBAL_LOAD_ASYNC_TO_LDS_B128 (ASYNCcnt, per-lane),
// B tiles staged with the Tensor Data Mover (TENSORcnt, wave 0), using D#
// pad_enable (4 dwords after every 16 dwords) to realize the LDS row stride
// of 40 halves in hardware.  Double-buffered in LDS.
__global__ __launch_bounds__(256) void gemm_f16_kernel(
    const _Float16* __restrict__ A, const _Float16* __restrict__ Bw,
    float* __restrict__ C, const float* __restrict__ bias,
    int Kp, int Np, int act, int beta)
{
  constexpr int LDT = 40;  // LDS row stride in halves (64B data + 16B pad)
  __shared__ __align__(16) _Float16 As[2][64 * LDT];
  __shared__ __align__(16) _Float16 Bs[2][64 * LDT];

  const int tid  = threadIdx.x;
  const int lane = tid & 31;
  const int wave = tid >> 5;       // 0..7
  const int tr   = wave >> 1;      // 16-row tile 0..3
  const int tc0  = (wave & 1) * 2; // 16-col tiles tc0, tc0+1
  const int m0   = blockIdx.x * 64;
  const int n0   = blockIdx.y * 64;
  const int mloc = lane & 15;
  const int hi   = lane >> 4;
  const int crow = tid >> 2;        // 0..63 cooperative tile row
  const int cchk = (tid & 3) * 8;   // half-offset 0,8,16,24

  v8f acc0 = {}, acc1 = {};
  const int nk = Kp >> 5;

  auto stage = [&](int kt, int buf) {
    // ---- A tile: per-lane async 16B copy, global -> LDS (one op per wave) ----
    unsigned la = (unsigned)(unsigned long long)&As[buf][crow * LDT + cchk];
    const _Float16* ga = &A[(size_t)(m0 + crow) * Kp + (kt << 5) + cchk];
    asm volatile("global_load_async_to_lds_b128 %0, %1, off"
                 :: "v"(la), "v"(ga) : "memory");
#if HAS_TDM
    // ---- B tile: TDM DMA of a 64x32 f16 tile with LDS padding ----
    if (wave == 0) {
      unsigned long long gb =
          (unsigned long long)&Bw[(size_t)n0 * Kp + (kt << 5)];
      unsigned lb = (unsigned)(unsigned long long)&Bs[buf][0];
      v4u g0; v8i g1; v4i g2 = {0, 0, 0, 0}, g3 = {0, 0, 0, 0};
      v8i g4 = {0, 0, 0, 0, 0, 0, 0, 0};
      g0[0] = 1u;                                       // count=1, user mode
      g0[1] = __builtin_amdgcn_readfirstlane(lb);       // lds_addr (bytes)
      g0[2] = __builtin_amdgcn_readfirstlane((unsigned)gb);        // ga[31:0]
      g0[3] = __builtin_amdgcn_readfirstlane((unsigned)(gb >> 32)) // ga[56:32]
              | (2u << 30);                             // type=2 ("image")
      // data_size=2B(1), pad_enable, pad_interval=16dw(3), pad_amount=4dw(3)
      g1[0] = (1 << 16) | (1 << 20) | (3 << 22) | (3 << 25);
      g1[1] = 32u << 16;                    // tensor_dim0 = 32  (bits 79:48)
      g1[2] = 64u << 16;                    // tensor_dim1 = 64  (bits 111:80)
      g1[3] = 32u << 16;                    // tile_dim0  = 32   (bits 127:112)
      g1[4] = 64;                           // tile_dim1  = 64   (bits 143:128)
      g1[5] = __builtin_amdgcn_readfirstlane(Kp);  // dim0_stride (elements)
      g1[6] = 0; g1[7] = 0;
      __builtin_amdgcn_tensor_load_to_lds(g0, g1, g2, g3, g4, 0);
    }
#else
    // fallback: B staged like A via async copies
    unsigned lbp = (unsigned)(unsigned long long)&Bs[buf][crow * LDT + cchk];
    const _Float16* gbp = &Bw[(size_t)(n0 + crow) * Kp + (kt << 5) + cchk];
    asm volatile("global_load_async_to_lds_b128 %0, %1, off"
                 :: "v"(lbp), "v"(gbp) : "memory");
#endif
  };

  stage(0, 0);
  for (int kt = 0; kt < nk; ++kt) {
    const int buf = kt & 1;
    if (kt + 1 < nk) {
      stage(kt + 1, buf ^ 1);  // prefetch next tile into other LDS buffer
#if HAS_TDM
      asm volatile("s_wait_asynccnt 0x1" ::: "memory");
      __builtin_amdgcn_s_wait_tensorcnt(1);
#else
      asm volatile("s_wait_asynccnt 0x2" ::: "memory");
#endif
    } else {
      asm volatile("s_wait_asynccnt 0x0" ::: "memory");
#if HAS_TDM
      __builtin_amdgcn_s_wait_tensorcnt(0);
#endif
    }
    __syncthreads();

    // A fragment: lane (hi,m) holds K = {8*hi..8*hi+7} and {16+8*hi..23+8*hi}
    const _Float16* ap = &As[buf][(tr * 16 + mloc) * LDT + 8 * hi];
    v16h a = cat8(*(const v8h*)ap, *(const v8h*)(ap + 16));
    // B fragment: lane (hi,n) holds K = {8*hi..8*hi+7} and {16+8*hi..23+8*hi} of column n
    const _Float16* bp0 = &Bs[buf][(tc0 * 16 + mloc) * LDT + 8 * hi];
    v16h b0 = cat8(*(const v8h*)bp0, *(const v8h*)(bp0 + 16));
    const _Float16* bp1 = bp0 + 16 * LDT;
    v16h b1 = cat8(*(const v8h*)bp1, *(const v8h*)(bp1 + 16));

    acc0 = __builtin_amdgcn_wmma_f32_16x16x32_f16(false, a, false, b0, (short)0, acc0, false, false);
    acc1 = __builtin_amdgcn_wmma_f32_16x16x32_f16(false, a, false, b1, (short)0, acc1, false, false);
    __syncthreads();
  }

#pragma unroll
  for (int r = 0; r < 8; ++r) {
    const int row = m0 + tr * 16 + r + 8 * hi;
#pragma unroll
    for (int t = 0; t < 2; ++t) {
      const int col = n0 + (tc0 + t) * 16 + mloc;
      float v = (t == 0) ? acc0[r] : acc1[r];
      if (bias) v += bias[col];
      if (act == 1) v = softplus_f(v);
      const size_t off = (size_t)row * Np + col;
      if (beta) v += C[off];
      C[off] = v;
    }
  }
}

// ---------------- weight staging (f32 -> zero-padded f16 [Np][Kp]) ----------------
__global__ void stage_w_kernel(const float* __restrict__ W, _Float16* __restrict__ dst,
                               int N, int K, int Np, int Kp)
{
  int t = blockIdx.x * 256 + threadIdx.x;
  if (t >= Np * Kp) return;
  int k = t % Kp, n = t / Kp;
  dst[t] = (n < N && k < K) ? (_Float16)W[n * K + k] : (_Float16)0.f;
}

__global__ void stage_bias_kernel(const float* __restrict__ b, float* __restrict__ dst, int N, int Np)
{
  int t = blockIdx.x * 256 + threadIdx.x;
  if (t >= Np) return;
  dst[t] = (t < N) ? b[t] : 0.f;
}

__global__ void zero_f32_kernel(float* p, int n)
{
  int t = blockIdx.x * 256 + threadIdx.x;
  if (t < n) p[t] = 0.f;
}

// ---------------- patch extraction: im2col for patch-embed GEMM ----------------
__global__ void patch_extract_kernel(const float* __restrict__ x, _Float16* __restrict__ Ap)
{
  size_t t = (size_t)blockIdx.x * 256 + threadIdx.x;  // M * 192
  if (t >= (size_t)M * 192) return;
  int k = (int)(t % 192); int row = (int)(t / 192);
  int b = row >> 10, pidx = row & 1023, i = pidx >> 5, j = pidx & 31;
  int c = k >> 6, rem = k & 63, p = rem >> 3, q = rem & 7;
  size_t xi = (((size_t)b * 3 + c) * 256 + (i * 8 + p)) * 256 + (j * 8 + q);
  Ap[t] = (_Float16)x[xi];
}

// ---------------- patch LN + silu + positional concat: tok -> h (f32 + padded f16) ----------------
__global__ __launch_bounds__(256) void pe_ln_kernel(
    const float* __restrict__ tok, const float* __restrict__ g, const float* __restrict__ bb,
    float* __restrict__ h32, _Float16* __restrict__ h16)
{
  int gw = blockIdx.x * 8 + (threadIdx.x >> 5);  // one wave per row
  int lane = threadIdx.x & 31;
  if (gw >= M) return;
  const float* trow = tok + (size_t)gw * E;
  float vals[6], s = 0.f, s2 = 0.f;
#pragma unroll
  for (int i = 0; i < 6; ++i) { float v = trow[lane + 32 * i]; vals[i] = v; s += v; s2 += v * v; }
  for (int off = 16; off; off >>= 1) { s += __shfl_xor(s, off, 32); s2 += __shfl_xor(s2, off, 32); }
  float mean = s / (float)E;
  float inv  = rsqrtf(s2 / (float)E - mean * mean + 1e-5f);
  float* hr = h32 + (size_t)gw * DM;
  _Float16* hf = h16 + (size_t)gw * KP_H;
#pragma unroll
  for (int i = 0; i < 6; ++i) {
    int e = lane + 32 * i;
    float v = (vals[i] - mean) * inv * g[e] + bb[e];
    v = silu_f(v);
    hr[e] = v; hf[e] = (_Float16)v;
  }
  int l = gw & 1023;
  if (lane < 4) {
    const float scale = 6.28318530717958647692f / 32.f;
    float pos = (float)l * scale;
    float rw  = (float)(l >> 5) * scale;
    float cp = (lane == 0) ? __sinf(pos) : (lane == 1) ? __cosf(pos)
             : (lane == 2) ? __sinf(rw)  : __cosf(rw);
    hr[E + lane] = cp; hf[E + lane] = (_Float16)cp;
  }
  if (lane < KP_H - DM) hf[DM + lane] = (_Float16)0.f;  // pad
}

// ---------------- depthwise causal / anti-causal conv + silu ----------------
__global__ void conv_silu_kernel(const float* __restrict__ xz, const float* __restrict__ cw,
                                 const float* __restrict__ cb, float* __restrict__ xc32,
                                 _Float16* __restrict__ xc16, int dir)
{
  size_t t = (size_t)blockIdx.x * 256 + threadIdx.x;  // M * KP_XC
  if (t >= (size_t)M * KP_XC) return;
  int d = (int)(t % KP_XC); int row = (int)(t / KP_XC);
  if (d >= DI) { xc16[t] = (_Float16)0.f; return; }
  int b = row >> 10, l = row & 1023;
  float acc = cb[d];
#pragma unroll
  for (int k = 0; k < 4; ++k) {
    int ls = dir ? (l + 3 - k) : (l - 3 + k);
    if (ls >= 0 && ls < L)
      acc += cw[d * 4 + k] * xz[(size_t)(b * L + ls) * NP_XZ + d];
  }
  float v = silu_f(acc);
  xc32[(size_t)row * DI + d] = v;
  xc16[t] = (_Float16)v;
}

// ---------------- extract dt columns of x_dbl into padded f16 A ----------------
__global__ void dt_extract_kernel(const float* __restrict__ xdbl, _Float16* __restrict__ dtA)
{
  size_t t = (size_t)blockIdx.x * 256 + threadIdx.x;  // M * KP_DT
  if (t >= (size_t)M * KP_DT) return;
  int k = (int)(t % KP_DT); int row = (int)(t / KP_DT);
  dtA[t] = (k < DTR) ? (_Float16)xdbl[(size_t)row * NP_XDBL + k] : (_Float16)0.f;
}

// ---------------- selective scan, fused with +D*xc and silu(z) gate ----------------
__global__ void scan_kernel(const float* __restrict__ delta, const float* __restrict__ xdbl,
                            const float* __restrict__ xc32, const float* __restrict__ xz,
                            const float* __restrict__ A_log, const float* __restrict__ Dp,
                            _Float16* __restrict__ y2, int dir)
{
  int t = blockIdx.x * 256 + threadIdx.x;  // BATCH * KP_XC
  if (t >= BATCH * KP_XC) return;
  int d = t % KP_XC, b = t / KP_XC;
  if (d >= DI) {
    for (int l = 0; l < L; ++l) y2[(size_t)(b * L + l) * KP_XC + d] = (_Float16)0.f;
    return;
  }
  float Ad[DS], h[DS];
#pragma unroll
  for (int n = 0; n < DS; ++n) { Ad[n] = -__expf(A_log[d * DS + n]); h[n] = 0.f; }
  const float Dd = Dp[d];
  for (int s = 0; s < L; ++s) {
    int l = dir ? (L - 1 - s) : s;
    size_t row = (size_t)b * L + l;
    float dv  = delta[row * NP_DLT + d];
    float xcv = xc32[row * DI + d];
    float zv  = xz[row * NP_XZ + DI + d];
    const float* bm = &xdbl[row * NP_XDBL + DTR];
    const float* cm = &xdbl[row * NP_XDBL + DTR + DS];
    float y = 0.f;
#pragma unroll
    for (int n = 0; n < DS; ++n) {
      h[n] = __expf(dv * Ad[n]) * h[n] + dv * bm[n] * xcv;
      y += h[n] * cm[n];
    }
    float yv = (y + Dd * xcv) * silu_f(zv);
    y2[row * KP_XC + d] = (_Float16)yv;
  }
}

// ---------------- final: 0.5*(fo+bo) -> blk LN -> silu -> +h -> batch-mean accum ----------------
__global__ __launch_bounds__(256) void final_row_kernel(
    const float* __restrict__ oacc, const float* __restrict__ h32,
    const float* __restrict__ g, const float* __restrict__ bb, float* __restrict__ bacc)
{
  int gw = blockIdx.x * 8 + (threadIdx.x >> 5);
  int lane = threadIdx.x & 31;
  if (gw >= M) return;
  const float* o = oacc + (size_t)gw * NP_OUT;
  float vals[7], s = 0.f, s2 = 0.f;
#pragma unroll
  for (int i = 0; i < 7; ++i) {
    int e = lane + 32 * i;
    float v = (e < DM) ? 0.5f * o[e] : 0.f;
    vals[i] = v; s += v; s2 += v * v;
  }
  for (int off = 16; off; off >>= 1) { s += __shfl_xor(s, off, 32); s2 += __shfl_xor(s2, off, 32); }
  float mean = s / (float)DM;
  float inv  = rsqrtf(s2 / (float)DM - mean * mean + 1e-5f);
  int b = gw >> 10;
  const float* hr = h32 + (size_t)gw * DM;
#pragma unroll
  for (int i = 0; i < 7; ++i) {
    int e = lane + 32 * i;
    if (e < DM) {
      float v = (vals[i] - mean) * inv * g[e] + bb[e];
      v = silu_f(v) + hr[e];
      atomicAdd(&bacc[b * DM + e], v);
    }
  }
}

__global__ void cls_kernel(const float* __restrict__ bacc, const float* __restrict__ cw,
                           const float* __restrict__ cb, float* __restrict__ out)
{
  int b = blockIdx.x, lane = threadIdx.x;  // 32 threads
  float s = 0.f;
  for (int e = lane; e < DM; e += 32) s += (bacc[b * DM + e] * (1.f / (float)L)) * cw[e];
  for (int off = 16; off; off >>= 1) s += __shfl_xor(s, off, 32);
  if (lane == 0) out[b] = s + cb[0];
}

// ---------------- launcher ----------------
static inline unsigned cdiv(size_t a, size_t b) { return (unsigned)((a + b - 1) / b); }

extern "C" void kernel_launch(void* const* d_in, const int* in_sizes, int n_in,
                              void* d_out, int out_size, void* d_ws, size_t ws_size,
                              hipStream_t stream)
{
  const float* x       = (const float*)d_in[0];
  const float* patch_w = (const float*)d_in[1];
  const float* patch_b = (const float*)d_in[2];
  const float* pe_g    = (const float*)d_in[3];
  const float* pe_b    = (const float*)d_in[4];
  const float* blk_g   = (const float*)d_in[5];
  const float* blk_b   = (const float*)d_in[6];
  const float* cls_w   = (const float*)d_in[7];
  const float* cls_b   = (const float*)d_in[8];
  const float* in_w[2]   = {(const float*)d_in[9],  (const float*)d_in[18]};
  const float* conv_w[2] = {(const float*)d_in[10], (const float*)d_in[19]};
  const float* conv_b[2] = {(const float*)d_in[11], (const float*)d_in[20]};
  const float* xp_w[2]   = {(const float*)d_in[12], (const float*)d_in[21]};
  const float* dt_w[2]   = {(const float*)d_in[13], (const float*)d_in[22]};
  const float* dt_b[2]   = {(const float*)d_in[14], (const float*)d_in[23]};
  const float* A_log[2]  = {(const float*)d_in[15], (const float*)d_in[24]};
  const float* Dp[2]     = {(const float*)d_in[16], (const float*)d_in[25]};
  const float* out_w[2]  = {(const float*)d_in[17], (const float*)d_in[26]};

  // workspace bump allocator
  char* ws = (char*)d_ws;
  size_t off = 0;
  auto alloc = [&](size_t bytes) -> char* {
    char* p = ws + off;
    off += (bytes + 255) & ~(size_t)255;
    return p;
  };
  float*    bacc     = (float*)alloc((size_t)BATCH * DM * 4);
  _Float16* patchW16 = (_Float16*)alloc((size_t)192 * 192 * 2);
  _Float16* inW16[2]  = {(_Float16*)alloc((size_t)NP_XZ * KP_H * 2),  (_Float16*)alloc((size_t)NP_XZ * KP_H * 2)};
  _Float16* xpW16[2]  = {(_Float16*)alloc((size_t)NP_XDBL * KP_XC * 2),(_Float16*)alloc((size_t)NP_XDBL * KP_XC * 2)};
  _Float16* dtW16[2]  = {(_Float16*)alloc((size_t)NP_DLT * KP_DT * 2), (_Float16*)alloc((size_t)NP_DLT * KP_DT * 2)};
  _Float16* outW16[2] = {(_Float16*)alloc((size_t)NP_OUT * KP_XC * 2), (_Float16*)alloc((size_t)NP_OUT * KP_XC * 2)};
  float*    dtB[2]    = {(float*)alloc((size_t)NP_DLT * 4), (float*)alloc((size_t)NP_DLT * 4)};
  _Float16* Apatch = (_Float16*)alloc((size_t)M * 192 * 2);
  float*    tok    = (float*)alloc((size_t)M * E * 4);
  float*    h32    = (float*)alloc((size_t)M * DM * 4);
  _Float16* h16    = (_Float16*)alloc((size_t)M * KP_H * 2);
  float*    oacc   = (float*)alloc((size_t)M * NP_OUT * 4);
  float*    xzb    = (float*)alloc((size_t)M * NP_XZ * 4);
  float*    xc32   = (float*)alloc((size_t)M * DI * 4);
  _Float16* xc16   = (_Float16*)alloc((size_t)M * KP_XC * 2);
  float*    xdbl   = (float*)alloc((size_t)M * NP_XDBL * 4);
  _Float16* dtA    = (_Float16*)alloc((size_t)M * KP_DT * 2);
  float*    delta  = (float*)alloc((size_t)M * NP_DLT * 4);
  _Float16* y2     = (_Float16*)alloc((size_t)M * KP_XC * 2);

  // ---- stage weights / zero accumulators ----
  zero_f32_kernel<<<cdiv(BATCH * DM, 256), 256, 0, stream>>>(bacc, BATCH * DM);
  stage_w_kernel<<<cdiv(192 * 192, 256), 256, 0, stream>>>(patch_w, patchW16, 192, 192, 192, 192);
  for (int d = 0; d < 2; ++d) {
    stage_w_kernel<<<cdiv(NP_XZ * KP_H, 256), 256, 0, stream>>>(in_w[d], inW16[d], 784, DM, NP_XZ, KP_H);
    stage_w_kernel<<<cdiv(NP_XDBL * KP_XC, 256), 256, 0, stream>>>(xp_w[d], xpW16[d], 45, DI, NP_XDBL, KP_XC);
    stage_w_kernel<<<cdiv(NP_DLT * KP_DT, 256), 256, 0, stream>>>(dt_w[d], dtW16[d], DI, DTR, NP_DLT, KP_DT);
    stage_w_kernel<<<cdiv(NP_OUT * KP_XC, 256), 256, 0, stream>>>(out_w[d], outW16[d], DM, DI, NP_OUT, KP_XC);
    stage_bias_kernel<<<cdiv(NP_DLT, 256), 256, 0, stream>>>(dt_b[d], dtB[d], DI, NP_DLT);
  }

  // ---- patch embed: im2col + WMMA GEMM (+bias), then LN+silu+pos -> h ----
  patch_extract_kernel<<<cdiv((size_t)M * 192, 256), 256, 0, stream>>>(x, Apatch);
  gemm_f16_kernel<<<dim3(M / 64, 192 / 64), 256, 0, stream>>>(
      Apatch, patchW16, tok, patch_b, 192, 192, 0, 0);
  pe_ln_kernel<<<M / 8, 256, 0, stream>>>(tok, pe_g, pe_b, h32, h16);

  // ---- two mamba directions (flip handled by conv taps + scan order) ----
  for (int dir = 0; dir < 2; ++dir) {
    gemm_f16_kernel<<<dim3(M / 64, NP_XZ / 64), 256, 0, stream>>>(
        h16, inW16[dir], xzb, nullptr, KP_H, NP_XZ, 0, 0);
    conv_silu_kernel<<<cdiv((size_t)M * KP_XC, 256), 256, 0, stream>>>(
        xzb, conv_w[dir], conv_b[dir], xc32, xc16, dir);
    gemm_f16_kernel<<<dim3(M / 64, NP_XDBL / 64), 256, 0, stream>>>(
        xc16, xpW16[dir], xdbl, nullptr, KP_XC, NP_XDBL, 0, 0);
    dt_extract_kernel<<<cdiv((size_t)M * KP_DT, 256), 256, 0, stream>>>(xdbl, dtA);
    gemm_f16_kernel<<<dim3(M / 64, NP_DLT / 64), 256, 0, stream>>>(
        dtA, dtW16[dir], delta, dtB[dir], KP_DT, NP_DLT, 1 /*softplus*/, 0);
    scan_kernel<<<cdiv(BATCH * KP_XC, 256), 256, 0, stream>>>(
        delta, xdbl, xc32, xzb, A_log[dir], Dp[dir], y2, dir);
    gemm_f16_kernel<<<dim3(M / 64, NP_OUT / 64), 256, 0, stream>>>(
        y2, outW16[dir], oacc, nullptr, KP_XC, NP_OUT, 0, dir /*beta: accumulate bo*/);
  }

  // ---- final: LN + silu + residual, batch mean, classifier ----
  final_row_kernel<<<M / 8, 256, 0, stream>>>(oacc, h32, blk_g, blk_b, bacc);
  cls_kernel<<<BATCH, 32, 0, stream>>>(bacc, cls_w, cls_b, (float*)d_out);
}
